// Encoder_55748675502365
// MI455X (gfx1250) — matrix operand order repaired and checked by reading, not compiled, not run on hardware
//
#include <hip/hip_runtime.h>
#include <cstdint>
#include <cstddef>

#define NN   50000          // nodes
#define NE   800000         // edges (before self loops)
#define KDIM 256            // GEMM K (in_dim and 2*HID)
#define HD   128            // hidden / output dim
#define WELEMS (KDIM * HD)  // elements per weight matrix

typedef __attribute__((ext_vector_type(16))) _Float16 v16h;
typedef __attribute__((ext_vector_type(8)))  _Float16 v8h;
typedef __attribute__((ext_vector_type(8)))  float    v8f;

// ---- order-preserving float <-> uint key (for atomicMax on floats) ----
__device__ __forceinline__ unsigned f2key(float f) {
    unsigned b = __float_as_uint(f);
    return (b & 0x80000000u) ? ~b : (b | 0x80000000u);
}
__device__ __forceinline__ float key2f(unsigned k) {
    unsigned b = (k & 0x80000000u) ? (k ^ 0x80000000u) : ~k;
    return __uint_as_float(b);
}
#define NEG_INF_KEY 0x007FFFFFu   // f2key(-inf)

// ---- pack W [K x 128] f32 -> Wt [128 x K] f16 (B-fragment friendly) ----
__global__ __launch_bounds__(256) void pack_w_kernel(const float* __restrict__ W,
                                                     _Float16* __restrict__ Wt) {
    int i = blockIdx.x * blockDim.x + threadIdx.x;
    if (i >= WELEMS) return;
    int k = i >> 7, n = i & (HD - 1);
    Wt[(size_t)n * KDIM + k] = (_Float16)W[(size_t)k * HD + n];
}

// ---- f32 -> f16 conversion, optional fused ReLU ----
__global__ __launch_bounds__(256) void cvt_f16_kernel(const float* __restrict__ x,
                                                      _Float16* __restrict__ y,
                                                      long n, int relu) {
    long i = (long)blockIdx.x * blockDim.x + threadIdx.x;
    if (i >= n) return;
    float v = x[i];
    if (relu) v = v > 0.f ? v : 0.f;
    y[i] = (_Float16)v;
}

// ---- WMMA GEMM: C[m, coff+n] = sum_k A[m,k] * Bt[n,k] (+ bias[n]) ----
// A: [M x K] f16 row-major, Bt: [128 x K] f16, C: f32 with row stride ldc.
// Block = 256 threads = 8 waves; wave w -> 16-column tile w; grid.x = M/16.
// Fragment layouts per CDNA5 ISA 7.12.2 (16-bit A 16x32, 16-bit B 32x16).
__global__ __launch_bounds__(256) void gemm_wmma_kernel(const _Float16* __restrict__ A,
                                                        const _Float16* __restrict__ Bt,
                                                        float* __restrict__ C,
                                                        int K, int ldc, int coff,
                                                        const float* __restrict__ bias) {
    const int m0   = blockIdx.x * 16;
    const int wave = threadIdx.x >> 5;
    const int lane = threadIdx.x & 31;
    const int half = lane >> 4;     // lane group 0-15 vs 16-31
    const int r    = lane & 15;
    const int n0   = wave * 16;

    const _Float16* arow = A  + (size_t)(m0 + r) * K;   // A row m = m0 + r
    const _Float16* brow = Bt + (size_t)(n0 + r) * K;   // B col n = n0 + r

    v8f acc = {};
    for (int k0 = 0; k0 < K; k0 += 32) {
        union { v16h v; v8h h[2]; } ua, ub;
        // A lane (group h): VGPR0-3 = K [k0+8h .. k0+8h+7], VGPR4-7 = K [k0+16+8h ..]
        ua.h[0] = *(const v8h*)(arow + k0 + half * 8);
        ua.h[1] = *(const v8h*)(arow + k0 + 16 + half * 8);
        // B lane (group h): VGPR0-3 = K [k0+8h .. k0+8h+7], VGPR4-7 = K [k0+16+8h ..]
        ub.h[0] = *(const v8h*)(brow + k0 + half * 8);
        ub.h[1] = *(const v8h*)(brow + k0 + 16 + half * 8);
        acc = __builtin_amdgcn_wmma_f32_16x16x32_f16(
            false, ua.v, false, ub.v, (short)0, acc, false, false);
    }

    const int col = n0 + r;
    const float bval = bias ? bias[col] : 0.f;
    #pragma unroll
    for (int v = 0; v < 8; ++v) {
        // C/D layout: lanes 0-15 -> M = v, lanes 16-31 -> M = v + 8; N = lane&15
        int row = m0 + v + half * 8;
        C[(size_t)row * ldc + coff + col] = acc[v] + bval;
    }
}

// ---- per-node attention scalars: alpha_s = h.a_src, alpha_d = h.a_dst ----
__global__ __launch_bounds__(256) void alpha_kernel(const float* __restrict__ h,
                                                    const float* __restrict__ av,
                                                    const float* __restrict__ aw,
                                                    float* __restrict__ outs,
                                                    float* __restrict__ outd) {
    int node = (int)((blockIdx.x * (unsigned)blockDim.x + threadIdx.x) >> 5);
    int lane = threadIdx.x & 31;
    if (node >= NN) return;
    const float* row = h + (size_t)node * HD;
    float s = 0.f, d = 0.f;
    #pragma unroll
    for (int j = 0; j < 4; ++j) {
        int c = lane + 32 * j;
        float v = row[c];
        s += v * av[c];
        d += v * aw[c];
    }
    #pragma unroll
    for (int o = 16; o > 0; o >>= 1) {
        s += __shfl_xor(s, o, 32);
        d += __shfl_xor(d, o, 32);
    }
    if (lane == 0) { outs[node] = s; outd[node] = d; }
}

// ---- seed accumulator with bias; reset max keys and denominators ----
__global__ __launch_bounds__(256) void init_conv_kernel(float* __restrict__ OS,
                                                        const float* __restrict__ bias,
                                                        unsigned* __restrict__ MX,
                                                        float* __restrict__ DN, int coff) {
    int i = blockIdx.x * blockDim.x + threadIdx.x;
    if (i >= NN * HD) return;
    int n = i >> 7, c = i & (HD - 1);
    OS[(size_t)n * (2 * HD) + coff + c] = bias[c];
    if (c == 0) { MX[n] = NEG_INF_KEY; DN[n] = 0.f; }
}

// ---- pass 1: edge logits (leaky relu) + per-target segment max ----
__global__ __launch_bounds__(256) void edge_logit_max_kernel(const int* __restrict__ src,
                                                             const int* __restrict__ dst,
                                                             const float* __restrict__ AS,
                                                             const float* __restrict__ AD,
                                                             float* __restrict__ EL,
                                                             unsigned* __restrict__ MX) {
    int stride = gridDim.x * blockDim.x;
    for (int e = blockIdx.x * blockDim.x + threadIdx.x; e < NE + NN; e += stride) {
        int s = e < NE ? src[e] : (e - NE);   // self loop tail
        int d = e < NE ? dst[e] : (e - NE);
        float v = AS[s] + AD[d];
        v = v >= 0.f ? v : 0.2f * v;          // leaky relu, slope 0.2
        EL[e] = v;
        atomicMax(&MX[d], f2key(v));
    }
}

// ---- pass 2: ex = exp(e - max[dst]); denom[dst] += ex ----
__global__ __launch_bounds__(256) void edge_expsum_kernel(const int* __restrict__ dst,
                                                          const float* __restrict__ EL,
                                                          const unsigned* __restrict__ MX,
                                                          float* __restrict__ EX,
                                                          float* __restrict__ DN) {
    int stride = gridDim.x * blockDim.x;
    for (int e = blockIdx.x * blockDim.x + threadIdx.x; e < NE + NN; e += stride) {
        int d = e < NE ? dst[e] : (e - NE);
        float ex = expf(EL[e] - key2f(MX[d]));
        EX[e] = ex;
        atomicAdd(&DN[d], ex);
    }
}

// ---- pass 3: OS[dst] += (ex/denom[dst]) * h[src]  (one wave per edge) ----
__global__ __launch_bounds__(256) void edge_aggregate_kernel(const int* __restrict__ src,
                                                             const int* __restrict__ dst,
                                                             const float* __restrict__ EX,
                                                             const float* __restrict__ DN,
                                                             const float* __restrict__ H,
                                                             float* __restrict__ OS, int coff) {
    int lane = threadIdx.x & 31;
    int gw = (int)((blockIdx.x * (unsigned)blockDim.x + threadIdx.x) >> 5);
    int nw = (gridDim.x * blockDim.x) >> 5;
    for (int e = gw; e < NE + NN; e += nw) {
        int s = e < NE ? src[e] : (e - NE);
        int d = e < NE ? dst[e] : (e - NE);
        float coef = EX[e] / DN[d];
        const float* hs = H + (size_t)s * HD;
        float* od = OS + (size_t)d * (2 * HD) + coff;
        #pragma unroll
        for (int j = 0; j < 4; ++j) {
            int c = lane + 32 * j;                  // coalesced 128B row gather
            atomicAdd(&od[c], coef * hs[c]);        // global_atomic_add_f32
        }
    }
}

extern "C" void kernel_launch(void* const* d_in, const int* in_sizes, int n_in,
                              void* d_out, int out_size, void* d_ws, size_t ws_size,
                              hipStream_t stream) {
    (void)in_sizes; (void)n_in; (void)out_size; (void)ws_size;
    const float* x      = (const float*)d_in[0];
    const int*   eo     = (const int*)d_in[1];
    const int*   es     = (const int*)d_in[2];
    const float* W_o1   = (const float*)d_in[3];
    const float* asr_o1 = (const float*)d_in[4];
    const float* ads_o1 = (const float*)d_in[5];
    const float* b_o1   = (const float*)d_in[6];
    const float* W_s1   = (const float*)d_in[7];
    const float* asr_s1 = (const float*)d_in[8];
    const float* ads_s1 = (const float*)d_in[9];
    const float* b_s1   = (const float*)d_in[10];
    const float* W_o2   = (const float*)d_in[11];
    const float* asr_o2 = (const float*)d_in[12];
    const float* ads_o2 = (const float*)d_in[13];
    const float* b_o2   = (const float*)d_in[14];
    const float* W_s2   = (const float*)d_in[15];
    const float* asr_s2 = (const float*)d_in[16];
    const float* ads_s2 = (const float*)d_in[17];
    const float* b_s2   = (const float*)d_in[18];
    const float* W_pr   = (const float*)d_in[19];
    const float* b_pr   = (const float*)d_in[20];

    // ---- workspace carving (every scratch byte written before read) ----
    uint8_t* base = (uint8_t*)d_ws;
    size_t off = 0;
    auto carve = [&](size_t bytes) -> void* {
        void* p = base + off;
        off += (bytes + 255) & ~(size_t)255;
        return p;
    };
    _Float16* F16 = (_Float16*)carve((size_t)NN * KDIM * sizeof(_Float16)); // x / x1 / x2 in f16
    _Float16* WT  = (_Float16*)carve((size_t)5 * WELEMS * sizeof(_Float16));
    float*    H   = (float*)carve((size_t)NN * HD * sizeof(float));         // per-conv h = x@W
    float*    OS  = (float*)carve((size_t)NN * 2 * HD * sizeof(float));     // [N x 256] fused concat
    float*    AS  = (float*)carve((size_t)NN * sizeof(float));
    float*    AD  = (float*)carve((size_t)NN * sizeof(float));
    unsigned* MX  = (unsigned*)carve((size_t)NN * sizeof(unsigned));
    float*    DN  = (float*)carve((size_t)NN * sizeof(float));
    float*    EL  = (float*)carve((size_t)(NE + NN) * sizeof(float));
    float*    EX  = (float*)carve((size_t)(NE + NN) * sizeof(float));

    _Float16* WT_o1 = WT + 0 * WELEMS;
    _Float16* WT_s1 = WT + 1 * WELEMS;
    _Float16* WT_o2 = WT + 2 * WELEMS;
    _Float16* WT_s2 = WT + 3 * WELEMS;
    _Float16* WT_pr = WT + 4 * WELEMS;

    dim3 b256(256);
    const int wBlocks = (WELEMS + 255) / 256;
    pack_w_kernel<<<wBlocks, b256, 0, stream>>>(W_o1, WT_o1);
    pack_w_kernel<<<wBlocks, b256, 0, stream>>>(W_s1, WT_s1);
    pack_w_kernel<<<wBlocks, b256, 0, stream>>>(W_o2, WT_o2);
    pack_w_kernel<<<wBlocks, b256, 0, stream>>>(W_s2, WT_s2);
    pack_w_kernel<<<wBlocks, b256, 0, stream>>>(W_pr, WT_pr);

    const long nx = (long)NN * KDIM;
    const int cvtBlocks = (int)((nx + 255) / 256);
    cvt_f16_kernel<<<cvtBlocks, b256, 0, stream>>>(x, F16, nx, 0);

    const int eBlocks = (NE + NN + 255) / 256;
    auto gat = [&](const int* edge, const _Float16* Wt, const float* av, const float* aw,
                   const float* bias, int coff) {
        gemm_wmma_kernel<<<NN / 16, b256, 0, stream>>>(F16, Wt, H, KDIM, HD, 0, nullptr);
        alpha_kernel<<<(NN + 7) / 8, b256, 0, stream>>>(H, av, aw, AS, AD);
        init_conv_kernel<<<(NN * HD + 255) / 256, b256, 0, stream>>>(OS, bias, MX, DN, coff);
        const int* src = edge;
        const int* dst = edge + NE;
        edge_logit_max_kernel<<<eBlocks, b256, 0, stream>>>(src, dst, AS, AD, EL, MX);
        edge_expsum_kernel<<<eBlocks, b256, 0, stream>>>(dst, EL, MX, EX, DN);
        edge_aggregate_kernel<<<4096, b256, 0, stream>>>(src, dst, EX, DN, H, OS, coff);
    };

    // layer 1: two branches write disjoint column halves of OS (free concat)
    gat(eo, WT_o1, asr_o1, ads_o1, b_o1, 0);
    gat(es, WT_s1, asr_s1, ads_s1, b_s1, HD);
    // relu(x1_os) -> f16 (x no longer needed; reuse F16)
    cvt_f16_kernel<<<cvtBlocks, b256, 0, stream>>>(OS, F16, nx, 1);
    // layer 2
    gat(eo, WT_o2, asr_o2, ads_o2, b_o2, 0);
    gat(es, WT_s2, asr_s2, ads_s2, b_s2, HD);
    // x2_os -> f16
    cvt_f16_kernel<<<cvtBlocks, b256, 0, stream>>>(OS, F16, nx, 0);
    // readout: x2_os @ W_pred + b_pred -> d_out [N x 128]
    gemm_wmma_kernel<<<NN / 16, b256, 0, stream>>>(F16, WT_pr, (float*)d_out,
                                                   KDIM, HD, 0, b_pr);
}
